// CausalSelfAttention_49572512531013
// MI455X (gfx1250) — matrix operand order repaired and checked by reading, not compiled, not run on hardware
//
#include <hip/hip_runtime.h>

#define BB 2
#define T  2048
#define CC 1024
#define NH 16
#define HD 64
#define NQKV 3072
#define MROWS (BB*T)   // 4096

typedef __attribute__((ext_vector_type(16))) __bf16 v16bf;
typedef __attribute__((ext_vector_type(8)))  float  v8f;

union ABFrag { v16bf v; unsigned int u[8]; };

__device__ __forceinline__ unsigned short f2bf(float f) {
    unsigned int u = __float_as_uint(f);
    unsigned int r = u + 0x7FFFu + ((u >> 16) & 1u);   // round-to-nearest-even
    return (unsigned short)(r >> 16);
}

// ---------------- pack / transpose kernels ----------------

__global__ void pack_bf16_k(const float* __restrict__ in, unsigned short* __restrict__ out, int n) {
    for (int i = blockIdx.x * blockDim.x + threadIdx.x; i < n; i += gridDim.x * blockDim.x)
        out[i] = f2bf(in[i]);
}

// in: [K][N] row-major fp32  ->  out: [N][K] bf16
__global__ void pack_transpose_k(const float* __restrict__ in, unsigned short* __restrict__ out,
                                 int K, int N) {
    int total = K * N;
    for (int i = blockIdx.x * blockDim.x + threadIdx.x; i < total; i += gridDim.x * blockDim.x) {
        int nn = i / K, kk = i % K;
        out[i] = f2bf(in[kk * N + nn]);
    }
}

// ---------------- QKV GEMM: [4096,1024]bf16 @ [1024,3072] + bias -> Q,K (b,h,t,hd) V^T (b,h,hd,t)

__global__ __launch_bounds__(256) void qkv_gemm_k(const unsigned short* __restrict__ Xb,
                                                  const unsigned short* __restrict__ Wt,   // [3072][1024]
                                                  const float* __restrict__ bias,
                                                  unsigned short* __restrict__ Q,
                                                  unsigned short* __restrict__ Kq,
                                                  unsigned short* __restrict__ Vt) {
    const int lane = threadIdx.x & 31, wid = threadIdx.x >> 5;
    const int wave = blockIdx.x * 8 + wid;
    const int NT = NQKV / 64;                     // 48 n-tiles
    const int mt = wave / NT, nt = wave % NT;
    const int m0 = mt * 32, n0 = nt * 64;
    const int lh = lane >> 4, ln = lane & 15;

    v8f acc[2][4];
#pragma unroll
    for (int mi = 0; mi < 2; ++mi)
#pragma unroll
        for (int ni = 0; ni < 4; ++ni)
#pragma unroll
            for (int r = 0; r < 8; ++r) acc[mi][ni][r] = 0.f;

    for (int k0 = 0; k0 < CC; k0 += 32) {
        ABFrag a[2];
#pragma unroll
        for (int mi = 0; mi < 2; ++mi) {
            int row = m0 + mi * 16 + ln;
#pragma unroll
            for (int v = 0; v < 8; ++v) {
                int g = v >> 2, vv = v & 3;
                a[mi].u[v] = *(const unsigned int*)(Xb + (size_t)row * CC + k0 + g * 16 + lh * 8 + vv * 2);
            }
        }
        ABFrag bfr[4];
#pragma unroll
        for (int ni = 0; ni < 4; ++ni) {
            int n = n0 + ni * 16 + ln;
#pragma unroll
            for (int v = 0; v < 8; ++v)
                bfr[ni].u[v] = *(const unsigned int*)(Wt + (size_t)n * CC + k0 + (v >> 2) * 16 + lh * 8 + (v & 3) * 2);
        }
#pragma unroll
        for (int mi = 0; mi < 2; ++mi)
#pragma unroll
            for (int ni = 0; ni < 4; ++ni)
                acc[mi][ni] = __builtin_amdgcn_wmma_f32_16x16x32_bf16(
                    false, a[mi].v, false, bfr[ni].v, (short)0, acc[mi][ni], false, false);
    }

#pragma unroll
    for (int mi = 0; mi < 2; ++mi)
#pragma unroll
        for (int ni = 0; ni < 4; ++ni) {
            int col = n0 + ni * 16 + ln;
            float bv = bias[col];
#pragma unroll
            for (int r = 0; r < 8; ++r) {
                int row = m0 + mi * 16 + r + lh * 8;
                int b = row / T, t = row % T;
                unsigned short h = f2bf(acc[mi][ni][r] + bv);
                if (col < CC) {
                    int hh = col / HD, d = col % HD;
                    Q[((size_t)(b * NH + hh) * T + t) * HD + d] = h;
                } else if (col < 2 * CC) {
                    int ch = col - CC, hh = ch / HD, d = ch % HD;
                    Kq[((size_t)(b * NH + hh) * T + t) * HD + d] = h;
                } else {
                    int ch = col - 2 * CC, hh = ch / HD, d = ch % HD;
                    Vt[((size_t)(b * NH + hh) * HD + d) * T + t] = h;
                }
            }
        }
}

// ---------------- flash attention: 1 wave per 16-query tile ----------------

__global__ __launch_bounds__(32) void flash_attn_k(const unsigned short* __restrict__ Q,
                                                   const unsigned short* __restrict__ Kq,
                                                   const unsigned short* __restrict__ Vt,
                                                   unsigned short* __restrict__ Yb) {
    __shared__ unsigned short pls[16 * 40];
    const int lane = threadIdx.x;
    const int lh = lane >> 4, ln = lane & 15;
    const int nqt = T / 16;
    const int bh = blockIdx.x / nqt;
    const int q0 = (blockIdx.x % nqt) * 16;
    const int b = bh / NH, hh = bh % NH;

    const unsigned short* Qh = Q  + (size_t)bh * T * HD;
    const unsigned short* Kh = Kq + (size_t)bh * T * HD;
    const unsigned short* Vh = Vt + (size_t)bh * HD * T;

    // Q tile as two A-fragments (hd chunks of 32)
    ABFrag qa[2];
#pragma unroll
    for (int c = 0; c < 2; ++c)
#pragma unroll
        for (int v = 0; v < 8; ++v) {
            int g = v >> 2, vv = v & 3;
            int k = c * 32 + g * 16 + lh * 8 + vv * 2;
            qa[c].u[v] = *(const unsigned int*)(Qh + (size_t)(q0 + ln) * HD + k);
        }

    float mrow[8], lrow[8];
    v8f yacc[4];
#pragma unroll
    for (int r = 0; r < 8; ++r) { mrow[r] = -3.0e38f; lrow[r] = 0.f; }
#pragma unroll
    for (int n = 0; n < 4; ++n)
#pragma unroll
        for (int r = 0; r < 8; ++r) yacc[n][r] = 0.f;

    const float scale = 0.125f;   // 1/sqrt(64)

    for (int j0 = 0; j0 <= q0; j0 += 32) {
        // ---- S = Q K^T   (16 q x 32 keys), two C-fragments
        v8f s[2];
#pragma unroll
        for (int f = 0; f < 2; ++f) {
#pragma unroll
            for (int r = 0; r < 8; ++r) s[f][r] = 0.f;
            int key = j0 + f * 16 + ln;
            int keyc = key < T ? key : (T - 1);
#pragma unroll
            for (int c = 0; c < 2; ++c) {
                ABFrag kb;
#pragma unroll
                for (int v = 0; v < 8; ++v) {
                    int kd = c * 32 + (v >> 2) * 16 + lh * 8 + (v & 3) * 2;
                    kb.u[v] = *(const unsigned int*)(Kh + (size_t)keyc * HD + kd);
                }
                s[f] = __builtin_amdgcn_wmma_f32_16x16x32_bf16(
                    false, qa[c].v, false, kb.v, (short)0, s[f], false, false);
            }
        }
        // ---- scale + causal mask (matches reference: mask fill -1e9 after scale)
#pragma unroll
        for (int f = 0; f < 2; ++f)
#pragma unroll
            for (int r = 0; r < 8; ++r) {
                int row = q0 + r + lh * 8;
                int key = j0 + f * 16 + ln;
                float sv = s[f][r] * scale;
                s[f][r] = (key <= row) ? sv : -1.0e9f;
            }
        // ---- online softmax: row max over the 32-key tile
        float mnew[8], corr[8];
#pragma unroll
        for (int r = 0; r < 8; ++r) {
            float t = fmaxf(s[0][r], s[1][r]);
#pragma unroll
            for (int off = 1; off < 16; off <<= 1) t = fmaxf(t, __shfl_xor(t, off, 32));
            mnew[r] = fmaxf(mrow[r], t);
            corr[r] = __expf(mrow[r] - mnew[r]);
            mrow[r] = mnew[r];
        }
        // ---- P = exp(S - m); write to LDS as bf16 16x32 (row-major, stride 40)
#pragma unroll
        for (int f = 0; f < 2; ++f)
#pragma unroll
            for (int r = 0; r < 8; ++r) {
                float p = __expf(s[f][r] - mnew[r]);
                lrow[r] = (f == 0 ? lrow[r] * corr[r] : lrow[r]) + p;
                pls[(r + lh * 8) * 40 + f * 16 + ln] = f2bf(p);
            }
#pragma unroll
        for (int n = 0; n < 4; ++n)
#pragma unroll
            for (int r = 0; r < 8; ++r) yacc[n][r] *= corr[r];
        __syncthreads();
        // ---- re-load P in A-fragment layout
        ABFrag pa;
#pragma unroll
        for (int v = 0; v < 8; ++v) {
            int g = v >> 2, vv = v & 3;
            int k = g * 16 + lh * 8 + vv * 2;
            pa.u[v] = *(const unsigned int*)(&pls[ln * 40 + k]);
        }
        __syncthreads();
        // ---- Y += P @ V   (V stored transposed: contiguous bf16 pairs along keys)
#pragma unroll
        for (int n = 0; n < 4; ++n) {
            ABFrag vb;
#pragma unroll
            for (int v = 0; v < 8; ++v) {
                int key = j0 + (v >> 2) * 16 + lh * 8 + (v & 3) * 2;
                int keyc = key < (T - 1) ? key : (T - 2);
                vb.u[v] = *(const unsigned int*)(Vh + (size_t)(n * 16 + ln) * T + keyc);
            }
            yacc[n] = __builtin_amdgcn_wmma_f32_16x16x32_bf16(
                false, pa.v, false, vb.v, (short)0, yacc[n], false, false);
        }
    }

    // ---- finalize: row-sum reduce l, normalize, store bf16 y [b,t,c]
#pragma unroll
    for (int r = 0; r < 8; ++r) {
        float t = lrow[r];
#pragma unroll
        for (int off = 1; off < 16; off <<= 1) t += __shfl_xor(t, off, 32);
        lrow[r] = t;
    }
#pragma unroll
    for (int n = 0; n < 4; ++n)
#pragma unroll
        for (int r = 0; r < 8; ++r) {
            int row = q0 + r + lh * 8;
            int col = hh * HD + n * 16 + ln;
            Yb[(size_t)(b * T + row) * CC + col] = f2bf(yacc[n][r] / lrow[r]);
        }
}

// ---------------- output projection: [4096,1024]bf16 @ [1024,1024] + bias -> fp32

__global__ __launch_bounds__(256) void proj_gemm_k(const unsigned short* __restrict__ Yb,
                                                   const unsigned short* __restrict__ Wt,  // [1024][1024] transposed
                                                   const float* __restrict__ bias,
                                                   float* __restrict__ out) {
    const int lane = threadIdx.x & 31, wid = threadIdx.x >> 5;
    const int wave = blockIdx.x * 8 + wid;
    const int NT = CC / 64;                       // 16 n-tiles
    const int mt = wave / NT, nt = wave % NT;
    const int m0 = mt * 32, n0 = nt * 64;
    const int lh = lane >> 4, ln = lane & 15;

    v8f acc[2][4];
#pragma unroll
    for (int mi = 0; mi < 2; ++mi)
#pragma unroll
        for (int ni = 0; ni < 4; ++ni)
#pragma unroll
            for (int r = 0; r < 8; ++r) acc[mi][ni][r] = 0.f;

    for (int k0 = 0; k0 < CC; k0 += 32) {
        ABFrag a[2];
#pragma unroll
        for (int mi = 0; mi < 2; ++mi) {
            int row = m0 + mi * 16 + ln;
#pragma unroll
            for (int v = 0; v < 8; ++v) {
                int g = v >> 2, vv = v & 3;
                a[mi].u[v] = *(const unsigned int*)(Yb + (size_t)row * CC + k0 + g * 16 + lh * 8 + vv * 2);
            }
        }
        ABFrag bfr[4];
#pragma unroll
        for (int ni = 0; ni < 4; ++ni) {
            int n = n0 + ni * 16 + ln;
#pragma unroll
            for (int v = 0; v < 8; ++v)
                bfr[ni].u[v] = *(const unsigned int*)(Wt + (size_t)n * CC + k0 + (v >> 2) * 16 + lh * 8 + (v & 3) * 2);
        }
#pragma unroll
        for (int mi = 0; mi < 2; ++mi)
#pragma unroll
            for (int ni = 0; ni < 4; ++ni)
                acc[mi][ni] = __builtin_amdgcn_wmma_f32_16x16x32_bf16(
                    false, a[mi].v, false, bfr[ni].v, (short)0, acc[mi][ni], false, false);
    }

#pragma unroll
    for (int mi = 0; mi < 2; ++mi)
#pragma unroll
        for (int ni = 0; ni < 4; ++ni) {
            int col = n0 + ni * 16 + ln;
            float bv = bias[col];
#pragma unroll
            for (int r = 0; r < 8; ++r) {
                int row = m0 + mi * 16 + r + lh * 8;
                out[(size_t)row * CC + col] = acc[mi][ni][r] + bv;
            }
        }
}

// ---------------- launcher ----------------

extern "C" void kernel_launch(void* const* d_in, const int* in_sizes, int n_in,
                              void* d_out, int out_size, void* d_ws, size_t ws_size,
                              hipStream_t stream) {
    const float* x      = (const float*)d_in[0];
    const float* w_attn = (const float*)d_in[1];
    const float* b_attn = (const float*)d_in[2];
    const float* w_proj = (const float*)d_in[3];
    const float* b_proj = (const float*)d_in[4];
    float* out = (float*)d_out;

    char* ws = (char*)d_ws;
    unsigned short* xb  = (unsigned short*)ws; ws += (size_t)MROWS * CC * 2;
    unsigned short* wat = (unsigned short*)ws; ws += (size_t)NQKV * CC * 2;
    unsigned short* wpt = (unsigned short*)ws; ws += (size_t)CC * CC * 2;
    unsigned short* Q   = (unsigned short*)ws; ws += (size_t)MROWS * CC * 2;
    unsigned short* K   = (unsigned short*)ws; ws += (size_t)MROWS * CC * 2;
    unsigned short* Vt  = (unsigned short*)ws; ws += (size_t)MROWS * CC * 2;
    unsigned short* Yb  = (unsigned short*)ws; ws += (size_t)MROWS * CC * 2;

    pack_bf16_k<<<1024, 256, 0, stream>>>(x, xb, MROWS * CC);
    pack_transpose_k<<<1024, 256, 0, stream>>>(w_attn, wat, CC, NQKV);
    pack_transpose_k<<<1024, 256, 0, stream>>>(w_proj, wpt, CC, CC);

    qkv_gemm_k<<<(MROWS / 32) * (NQKV / 64) / 8, 256, 0, stream>>>(xb, wat, b_attn, Q, K, Vt);

    flash_attn_k<<<BB * NH * (T / 16), 32, 0, stream>>>(Q, K, Vt, Yb);

    proj_gemm_k<<<(MROWS / 32) * (CC / 64) / 8, 256, 0, stream>>>(Yb, wpt, b_proj, out);
}
